// Model_6768868458809
// MI455X (gfx1250) — compile-verified, matrix operand order checked by reading, not run on hardware
//
#include <hip/hip_runtime.h>
#include <hip/hip_bf16.h>
#include <stdint.h>

// ---------------------------------------------------------------------------
// Problem constants (from the reference)
// ---------------------------------------------------------------------------
#define B_      8
#define MAXLEN_ 2048
#define EMB_    300
#define NF_     300
#define LP_     2046        // MAXLEN - NGRAM + 1

// Padded geometry for WMMA (16x16x32 f16 tiles)
#define EPAD    304         // EMB padded per n-gram slice (multiple of 16)
#define K1      928         // GEMM1 K: 3*304=912 -> pad to 29*32
#define K1T     29
#define EROWS   2052        // E16 rows per batch (2048 + 3 overrun + pad)
#define NFPAD   320         // NF padded to 20 n-tiles (10 pairs of 32)
#define K2      320         // GEMM2 K: NF=300 -> pad to 10*32
#define K2T     10
#define MPAD    2048        // LP padded to 128 m-tiles of 16
#define MT      128         // 16-row m-tiles per batch
#define MT32    64          // 32-row m-groups per batch (GEMM1)
#define NT1P    10          // 32-col n-groups (GEMM1)
#define NT2     128         // p-tiles (GEMM2)
#define WAVES2  8           // waves per block in GEMM2

typedef __attribute__((ext_vector_type(16))) _Float16     v16h;
typedef __attribute__((ext_vector_type(8)))  float        v8f;
typedef __attribute__((ext_vector_type(4)))  unsigned int v4u;

union Frag { v16h v; v4u q[2]; };

// Load a 16-half A/B fragment for v_wmma_f32_16x16x32_f16.
// Per ISA layout: lane L holds row (L&15); lanes 0-15 cover K=[0..7]+[16..23],
// lanes 16-31 cover K=[8..15]+[24..31] (relative to the k-tile).  With
// kofs = ktile*32 + ((lane>>4)&1)*8, element j<8 sits at rowbase[kofs+j] and
// element j>=8 at rowbase[kofs+16+(j-8)] -> two contiguous 16B loads.
__device__ __forceinline__ v16h load_frag(const _Float16* rowbase, int kofs) {
    Frag f;
    f.q[0] = *(const v4u*)(rowbase + kofs);
    f.q[1] = *(const v4u*)(rowbase + kofs + 16);
    return f.v;
}

// ---------------------------------------------------------------------------
// Prep kernels: f16 conversion with zero padding
// ---------------------------------------------------------------------------

// conv_w (NF,1,3,EMB) -> Wc[f][k], f in [0,320), k in [0,928); zeros in pads
__global__ void prep_conv_w(const float* __restrict__ conv_w, _Float16* __restrict__ Wc) {
    int idx = blockIdx.x * 256 + threadIdx.x;
    if (idx >= NFPAD * K1) return;
    int f = idx / K1, k = idx % K1;
    float v = 0.f;
    if (f < NF_ && k < 3 * EPAD) {
        int g = k / EPAD, e = k % EPAD;
        if (e < EMB_) v = conv_w[(size_t)f * (3 * EMB_) + g * EMB_ + e];
    }
    Wc[idx] = (_Float16)v;
}

// pa_w (LP,EMB) -> Pa16[p][k], p in [0,2048), k in [0,320); zeros in pads
__global__ void prep_pa_w(const float* __restrict__ pa_w, _Float16* __restrict__ Pa16) {
    int idx = blockIdx.x * 256 + threadIdx.x;
    if (idx >= MPAD * K2) return;
    int p = idx / K2, k = idx % K2;
    float v = (p < LP_ && k < EMB_) ? pa_w[(size_t)p * EMB_ + k] : 0.f;
    Pa16[idx] = (_Float16)v;
}

// Embedding gather: E16[b][l][e] = (f16) emb_table[x[b][l]][e], e<300; pad cols = 0
__global__ void gather_emb(const int* __restrict__ x, const float* __restrict__ emb,
                           _Float16* __restrict__ E16) {
    int idx = blockIdx.x * 256 + threadIdx.x;            // B*2048*304
    int e = idx % EPAD;
    int l = (idx / EPAD) % MAXLEN_;
    int b = idx / (EPAD * MAXLEN_);
    float v = 0.f;
    if (e < EMB_) {
        int tok = x[(size_t)b * MAXLEN_ + l];
        v = emb[(size_t)tok * EMB_ + e];
    }
    E16[(size_t)b * EROWS * EPAD + (size_t)l * EPAD + e] = (_Float16)v;
}

// ---------------------------------------------------------------------------
// GEMM1: U[b,l,f] = relu( sum_k im2col(E)[b,l,k] * Wc[f,k] + conv_b[f] )
// One wave per 32x32 output block (2x2 register tile of 16x16 WMMA tiles):
// per k-tile 4 fragment loads feed 4 WMMAs -> 1 KB/WMMA of L2 traffic.
// im2col is implicit: row (b,l) of A starts at E16 + (b*EROWS + l)*EPAD.
// ---------------------------------------------------------------------------
__global__ void __launch_bounds__(32)
gemm1_conv(const _Float16* __restrict__ E16, const _Float16* __restrict__ Wc,
           const float* __restrict__ conv_b, _Float16* __restrict__ U16) {
    int blk = blockIdx.x;            // b*MT32*NT1P + mgrp*NT1P + ngrp
    int ngrp = blk % NT1P;  blk /= NT1P;
    int mgrp = blk % MT32;
    int b    = blk / MT32;
    int lane = threadIdx.x;
    int hr   = lane & 15;
    int kb   = ((lane >> 4) & 1) * 8;

    int m0 = mgrp * 32;
    int f0 = ngrp * 32;
    const _Float16* arow0 = E16 + (size_t)b * EROWS * EPAD + (size_t)(m0 + hr) * EPAD;
    const _Float16* arow1 = arow0 + (size_t)16 * EPAD;
    const _Float16* brow0 = Wc + (size_t)(f0 + hr) * K1;
    const _Float16* brow1 = brow0 + (size_t)16 * K1;

    v8f acc00 = {}, acc01 = {}, acc10 = {}, acc11 = {};
#pragma unroll 1
    for (int kt = 0; kt < K1T; ++kt) {
        int kofs = kt * 32 + kb;
        v16h a0 = load_frag(arow0, kofs);
        v16h a1 = load_frag(arow1, kofs);
        v16h b0 = load_frag(brow0, kofs);
        v16h b1 = load_frag(brow1, kofs);
        acc00 = __builtin_amdgcn_wmma_f32_16x16x32_f16(false, a0, false, b0, (short)0, acc00, false, false);
        acc01 = __builtin_amdgcn_wmma_f32_16x16x32_f16(false, a0, false, b1, (short)0, acc01, false, false);
        acc10 = __builtin_amdgcn_wmma_f32_16x16x32_f16(false, a1, false, b0, (short)0, acc10, false, false);
        acc11 = __builtin_amdgcn_wmma_f32_16x16x32_f16(false, a1, false, b1, (short)0, acc11, false, false);
    }

    // D layout: lane holds column (hr), rows rbase..rbase+7 within the tile.
    int rsub = ((lane >> 4) & 1) * 8;
    _Float16* ubase = U16 + (size_t)b * MPAD * K2;
    v8f* accs[4] = { &acc00, &acc01, &acc10, &acc11 };
#pragma unroll
    for (int mi = 0; mi < 2; ++mi) {
#pragma unroll
        for (int ni = 0; ni < 2; ++ni) {
            v8f acc = *accs[mi * 2 + ni];
            int f = f0 + ni * 16 + hr;
            float cb = (f < NF_) ? conv_b[f] : 0.f;
            int rbase = m0 + mi * 16 + rsub;
            _Float16* out = ubase + (size_t)rbase * K2 + f;
#pragma unroll
            for (int r = 0; r < 8; ++r) {
                float v = acc[r] + cb;
                v = v > 0.f ? v : 0.f;
                out[(size_t)r * K2] = (_Float16)v;
            }
        }
    }
}

// ---------------------------------------------------------------------------
// GEMM2 + reduction epilogue:
//   logits[b,l] = sum_p tanh( U[b,l,:]·pa_w[p,:] + pa_b[p] ) * u_w[p]
// Block = 8 waves, each wave owns one 16-row m-tile with all 10 A-fragments
// register-resident (80 VGPRs).  The 10KB Pa16 p-tile is staged once per
// block into LDS (coalesced b128 copy, barrier), then all 8 waves feed their
// WMMAs from LDS -> 8x less vector-memory traffic.  tanh(..)*u_w reduced in
// registers, cross-lane shfl reduce, single store per row. No atomics.
// ---------------------------------------------------------------------------
__global__ void __launch_bounds__(256)
gemm2_att(const _Float16* __restrict__ U16, const _Float16* __restrict__ Pa16,
          const float* __restrict__ pa_b, const float* __restrict__ u_w,
          float* __restrict__ logits) {
    __shared__ __align__(16) _Float16 shB[16 * K2];       // 10240 B

    int tid  = threadIdx.x;
    int wave = tid >> 5, lane = tid & 31;
    int hr   = lane & 15;
    int kb   = ((lane >> 4) & 1) * 8;
    int b     = blockIdx.x >> 4;                          // grid = B * 16 blocks
    int mtile = (blockIdx.x & 15) * WAVES2 + wave;

    const _Float16* arow = U16 + (size_t)b * MPAD * K2 + (size_t)(mtile * 16 + hr) * K2;

    v16h af[K2T];
#pragma unroll
    for (int kt = 0; kt < K2T; ++kt) af[kt] = load_frag(arow, kt * 32 + kb);

    float sacc[8];
#pragma unroll
    for (int r = 0; r < 8; ++r) sacc[r] = 0.f;

    v4u* dst128 = (v4u*)shB;
#pragma unroll 1
    for (int nt = 0; nt < NT2; ++nt) {
        // Stage the contiguous 10KB p-tile (rows nt*16..nt*16+15 of Pa16) in LDS.
        const v4u* tsrc = (const v4u*)(Pa16 + (size_t)nt * 16 * K2);
        __syncthreads();                                  // previous tile consumed
        for (int i = tid; i < 640; i += 256) dst128[i] = tsrc[i];
        __syncthreads();                                  // tile visible to all waves

        // Pull the next tile toward L2 while this one is consumed.
        if (nt + 1 < NT2 && tid < 10)
            __builtin_prefetch(Pa16 + (size_t)(nt + 1) * 16 * K2 + (size_t)tid * 512, 0, 1);

        const _Float16* brow = shB + hr * K2;
        v8f acc = {};
#pragma unroll
        for (int kt = 0; kt < K2T; ++kt) {
            v16h bf = load_frag(brow, kt * 32 + kb);
            acc = __builtin_amdgcn_wmma_f32_16x16x32_f16(false, af[kt], false, bf,
                                                         (short)0, acc, false, false);
        }
        // D layout: this lane holds column p, rows rbase..rbase+7
        int p = nt * 16 + hr;
        float pb = (p < LP_) ? pa_b[p] : 0.f;
        float uw = (p < LP_) ? u_w[p] : 0.f;   // zero pad => padded p contribute 0
#pragma unroll
        for (int r = 0; r < 8; ++r) sacc[r] += tanhf(acc[r] + pb) * uw;
    }

    // Reduce over the 16 columns held by each half-wave (xor stays in-half)
#pragma unroll
    for (int off = 1; off < 16; off <<= 1) {
#pragma unroll
        for (int r = 0; r < 8; ++r) sacc[r] += __shfl_xor(sacc[r], off, 32);
    }

    if (hr == 0) {
        int rbase = mtile * 16 + ((lane >> 4) & 1) * 8;
        float* out = logits + (size_t)b * MPAD + rbase;
#pragma unroll
        for (int r = 0; r < 8; ++r) out[r] = sacc[r];
    }
}

// ---------------------------------------------------------------------------
// Finalize (one block per batch):
//   a = log_softmax(logits[0:2046]); f1=a*(U·w1); f2=a*(U·w2)
//   max_row[i] = f1[i] + b_sa + (i==argmax f2 ? 2ndmax f2 : max f2)
//   att = softmax(max_row);  f_a = sum_l att[l]*a[l]*U[l,:]
//   out[b,:] = f_a @ lin_w.T + lin_b
// ---------------------------------------------------------------------------
__global__ void __launch_bounds__(256)
finalize(const _Float16* __restrict__ U16, const float* __restrict__ logits,
         const float* __restrict__ w1, const float* __restrict__ w2,
         const float* __restrict__ b_sa, const float* __restrict__ lin_w,
         const float* __restrict__ lin_b,
         float* __restrict__ f1buf, float* __restrict__ f2buf,
         float* __restrict__ out) {
    const int b = blockIdx.x, tid = threadIdx.x;
    __shared__ float red[256];
    __shared__ int   redi[256];
    __shared__ float wbuf[MPAD];     // a[l], later att[l]*a[l]
    __shared__ float fa[EMB_ + 4];
    __shared__ float sh_m, sh_lse, sh_gmax, sh_smax, sh_m2, sh_s2;
    __shared__ int   sh_arg;

    const float* lg = logits + (size_t)b * MPAD;
    const _Float16* Ub = U16 + (size_t)b * MPAD * K2;

    // --- log_softmax over l < LP_ ---
    float m = -__builtin_inff();
    for (int l = tid; l < LP_; l += 256) m = fmaxf(m, lg[l]);
    red[tid] = m; __syncthreads();
    for (int s = 128; s > 0; s >>= 1) {
        if (tid < s) red[tid] = fmaxf(red[tid], red[tid + s]);
        __syncthreads();
    }
    if (tid == 0) sh_m = red[0];
    __syncthreads();
    float mm = sh_m;
    float se = 0.f;
    for (int l = tid; l < LP_; l += 256) se += __expf(lg[l] - mm);
    red[tid] = se; __syncthreads();
    for (int s = 128; s > 0; s >>= 1) {
        if (tid < s) red[tid] += red[tid + s];
        __syncthreads();
    }
    if (tid == 0) sh_lse = __logf(red[0]);
    __syncthreads();
    float lse = sh_lse;

    // --- f1 = a*(U·w1), f2 = a*(U·w2) ---
    for (int l = tid; l < LP_; l += 256) {
        const _Float16* ur = Ub + (size_t)l * K2;
        float d1 = 0.f, d2 = 0.f;
        for (int f = 0; f < NF_; ++f) {
            float u = (float)ur[f];
            d1 += u * w1[f];
            d2 += u * w2[f];
        }
        float a = lg[l] - mm - lse;
        wbuf[l] = a;
        f1buf[(size_t)b * MPAD + l] = a * d1;
        f2buf[(size_t)b * MPAD + l] = a * d2;
    }
    __syncthreads();

    // --- max & argmax of f2 ---
    float gm = -__builtin_inff(); int ga = 0;
    for (int l = tid; l < LP_; l += 256) {
        float v = f2buf[(size_t)b * MPAD + l];
        if (v > gm) { gm = v; ga = l; }
    }
    red[tid] = gm; redi[tid] = ga; __syncthreads();
    for (int s = 128; s > 0; s >>= 1) {
        if (tid < s && red[tid + s] > red[tid]) { red[tid] = red[tid + s]; redi[tid] = redi[tid + s]; }
        __syncthreads();
    }
    if (tid == 0) { sh_gmax = red[0]; sh_arg = redi[0]; }
    __syncthreads();
    // --- second max (excluding argmax) ---
    float sm = -__builtin_inff();
    for (int l = tid; l < LP_; l += 256)
        if (l != sh_arg) sm = fmaxf(sm, f2buf[(size_t)b * MPAD + l]);
    red[tid] = sm; __syncthreads();
    for (int s = 128; s > 0; s >>= 1) {
        if (tid < s) red[tid] = fmaxf(red[tid], red[tid + s]);
        __syncthreads();
    }
    if (tid == 0) sh_smax = red[0];
    __syncthreads();

    // --- max_row & softmax over i ---
    float bsa = b_sa[0];
    float rm = -__builtin_inff();
    for (int i = tid; i < LP_; i += 256) {
        float mr = f1buf[(size_t)b * MPAD + i] + bsa + ((i == sh_arg) ? sh_smax : sh_gmax);
        f1buf[(size_t)b * MPAD + i] = mr;   // reuse buffer for max_row
        rm = fmaxf(rm, mr);
    }
    red[tid] = rm; __syncthreads();
    for (int s = 128; s > 0; s >>= 1) {
        if (tid < s) red[tid] = fmaxf(red[tid], red[tid + s]);
        __syncthreads();
    }
    if (tid == 0) sh_m2 = red[0];
    __syncthreads();
    float m2 = sh_m2;
    float s2 = 0.f;
    for (int i = tid; i < LP_; i += 256) s2 += __expf(f1buf[(size_t)b * MPAD + i] - m2);
    red[tid] = s2; __syncthreads();
    for (int s = 128; s > 0; s >>= 1) {
        if (tid < s) red[tid] += red[tid + s];
        __syncthreads();
    }
    if (tid == 0) sh_s2 = red[0];
    __syncthreads();
    float inv_s2 = 1.f / sh_s2;
    for (int i = tid; i < LP_; i += 256) {
        float att = __expf(f1buf[(size_t)b * MPAD + i] - m2) * inv_s2;
        wbuf[i] = att * wbuf[i];            // att[l] * a[l]
    }
    __syncthreads();

    // --- f_a[f] = sum_l wbuf[l] * U[l,f] ---
    for (int f = tid; f < NF_; f += 256) {
        float s = 0.f;
        for (int l = 0; l < LP_; ++l) s += wbuf[l] * (float)Ub[(size_t)l * K2 + f];
        fa[f] = s;
    }
    __syncthreads();

    // --- out[b,c] = fa · lin_w[c,:] + lin_b[c] ---
    if (tid < 2) {
        float s = lin_b[tid];
        for (int f = 0; f < NF_; ++f) s += fa[f] * lin_w[(size_t)tid * NF_ + f];
        out[(size_t)b * 2 + tid] = s;
    }
}

// ---------------------------------------------------------------------------
// Launch
// ---------------------------------------------------------------------------
extern "C" void kernel_launch(void* const* d_in, const int* in_sizes, int n_in,
                              void* d_out, int out_size, void* d_ws, size_t ws_size,
                              hipStream_t stream) {
    const int*   x_batch   = (const int*)  d_in[0];
    const float* emb_table = (const float*)d_in[1];
    const float* conv_w    = (const float*)d_in[2];
    const float* conv_b    = (const float*)d_in[3];
    const float* pa_w      = (const float*)d_in[4];
    const float* pa_b      = (const float*)d_in[5];
    const float* u_w       = (const float*)d_in[6];
    const float* w1        = (const float*)d_in[7];
    const float* w2        = (const float*)d_in[8];
    const float* b_sa      = (const float*)d_in[9];
    const float* lin_w     = (const float*)d_in[10];
    const float* lin_b     = (const float*)d_in[11];
    float* out = (float*)d_out;

    // Workspace layout (all offsets naturally 256B aligned)
    char* ws = (char*)d_ws;
    size_t off = 0;
    _Float16* E16  = (_Float16*)(ws + off); off += (size_t)B_ * EROWS * EPAD * 2;   // 9.98 MB
    _Float16* Wc   = (_Float16*)(ws + off); off += (size_t)NFPAD * K1 * 2;          // 0.59 MB
    _Float16* Pa16 = (_Float16*)(ws + off); off += (size_t)MPAD * K2 * 2;           // 1.31 MB
    _Float16* U16  = (_Float16*)(ws + off); off += (size_t)B_ * MPAD * K2 * 2;      // 10.5 MB
    float* logits  = (float*)(ws + off);    off += (size_t)B_ * MPAD * 4;
    float* f1buf   = (float*)(ws + off);    off += (size_t)B_ * MPAD * 4;
    float* f2buf   = (float*)(ws + off);    off += (size_t)B_ * MPAD * 4;

    // Prep (f16 conversion + gather)
    prep_conv_w<<<(NFPAD * K1 + 255) / 256, 256, 0, stream>>>(conv_w, Wc);
    prep_pa_w  <<<(MPAD * K2 + 255) / 256, 256, 0, stream>>>(pa_w, Pa16);
    gather_emb <<<(B_ * MAXLEN_ * EPAD + 255) / 256, 256, 0, stream>>>(x_batch, emb_table, E16);

    // GEMM1: conv-as-GEMM, 2x2 register-tiled WMMA f16 -> U16
    gemm1_conv<<<B_ * MT32 * NT1P, 32, 0, stream>>>(E16, Wc, conv_b, U16);

    // GEMM2: attention logits, LDS-staged B, fused tanh/u_w reduction
    gemm2_att<<<B_ * (MT / WAVES2), 256, 0, stream>>>(U16, Pa16, pa_b, u_w, logits);

    // Epilogue: softmaxes, self-attention max trick, pooling, final linear
    finalize<<<B_, 256, 0, stream>>>(U16, logits, w1, w2, b_sa, lin_w, lin_b,
                                     f1buf, f2buf, out);

    (void)in_sizes; (void)n_in; (void)out_size; (void)ws_size;
}
